// Atttention_54348516164006
// MI455X (gfx1250) — matrix operand order repaired and checked by reading, not compiled, not run on hardware
//
#include <hip/hip_runtime.h>

// ---------------------------------------------------------------------------
// Problem constants (match reference)
// ---------------------------------------------------------------------------
#define BB 4
#define FF 2048
#define HH 16
#define DD 64
#define DH 1024           // hidden
#define MM (BB * FF)      // 8192 rows

typedef unsigned short u16;
typedef unsigned int   u32;
typedef unsigned long long u64;
typedef __attribute__((ext_vector_type(16))) __bf16 v16bf;
typedef __attribute__((ext_vector_type(8)))  float  v8f;
typedef __attribute__((ext_vector_type(4)))  u32    u32x4;
typedef __attribute__((ext_vector_type(8)))  int    i32x8;
typedef __attribute__((ext_vector_type(4)))  int    i32x4;
typedef __attribute__((ext_vector_type(4)))  float  f32x4;

union BFrag {
    v16bf bf;
    u32x4 q[2];
    u16   s[16];
};

__device__ __forceinline__ u16 f2bf(float f) {
    u32 u = __float_as_uint(f);
    u32 r = u + 0x7FFFu + ((u >> 16) & 1u);   // round-to-nearest-even
    return (u16)(r >> 16);
}

// Generic pointer to LDS: addr[31:0] is the LDS byte offset.
__device__ __forceinline__ u32 lds_off(const void* p) {
    return (u32)(uintptr_t)p;
}

// ---------------------------------------------------------------------------
// Tensor Data Mover: async-load a 2D bf16 tile (rows x rowlen, row pitch
// `stride` elements) from global memory into LDS at byte offset `ldsoff`.
// D# per CDNA5 ISA 8.3/8.4: group0 = {count|flags, lds_addr, global_addr,
// addr_hi|type=2}; group1 = {mask|data_size, dims/tiles/strides}.
// Tracked with TENSORcnt.  EXEC ignored; issue from one wave only.
// ---------------------------------------------------------------------------
__device__ __forceinline__ void tdm_load_2d(const u16* gsrc, u32 ldsoff,
                                            u32 rows, u32 rowlen, u32 stride)
{
    const u64 ga = (u64)(uintptr_t)gsrc;
    u32x4 g0;
    g0[0] = 1u;                                         // count=1, no gather
    g0[1] = ldsoff;                                     // lds_addr (bytes)
    g0[2] = (u32)(ga & 0xFFFFFFFFu);                    // global_addr[31:0]
    g0[3] = ((u32)(ga >> 32) & 0x01FFFFFFu)             // global_addr[56:32]
          | (2u << 30);                                 // type = 2 (image)
    const u32 td0 = stride;                             // tensor_dim0 (elems)
    const u32 td1 = rows;                               // tensor_dim1
    i32x8 g1;
    g1[0] = (int)(1u << 16);                            // wg_mask=0, data_size=1 (2B)
    g1[1] = (int)((td0 & 0xFFFFu) << 16);               // abar=0 | tensor_dim0 lo
    g1[2] = (int)((td0 >> 16) | ((td1 & 0xFFFFu) << 16));
    g1[3] = (int)((td1 >> 16) | (rowlen << 16));        // tile_dim0 = rowlen
    g1[4] = (int)(rows & 0xFFFFu);                      // tile_dim1 = rows, tile_dim2=0
    g1[5] = (int)stride;                                // tensor_dim0_stride lo32
    g1[6] = 0;                                          // stride hi | dim1_stride lo
    g1[7] = 0;
    i32x4 z4 = {};
    i32x8 z8 = {};
    __builtin_amdgcn_tensor_load_to_lds(g0, g1, z4, z4, z8, 0);
}

// ---------------------------------------------------------------------------
// Weight transpose + f32 -> bf16 convert:  Wt[n*DH + k] = bf16(W[k*DH + n])
// ---------------------------------------------------------------------------
__global__ __launch_bounds__(256) void transpose_w_kernel(const float* __restrict__ W,
                                                          u16* __restrict__ Wt)
{
    int i = blockIdx.x * 256 + threadIdx.x;   // 0 .. DH*DH-1
    int n = i >> 10;
    int k = i & (DH - 1);
    Wt[i] = f2bf(W[k * DH + n]);
}

// ---------------------------------------------------------------------------
// GEMM: C = A(8192 x 1024) @ W(1024 x 1024) + bias, via bf16 WMMA.
//   A_F32:    A is f32 row-major (converted to bf16 in-register), else bf16
//   OUT_MODE: 0 -> bf16 head-major [B][H][F][D]; 1 -> bf16 [B][H][D][F];
//             2 -> f32 row-major [M][DH]
// Block = 256 threads (8 waves), tile 128(M) x 64(N); wave tile 32x32.
// The block's Wt tile (64 rows x 32 K) is TDM-staged into LDS, double-buffered.
// ---------------------------------------------------------------------------
template<bool A_F32, int OUT_MODE>
__global__ __launch_bounds__(256) void gemm_kernel(const void* __restrict__ Aptr,
                                                   const u16* __restrict__ Wt,
                                                   const float* __restrict__ bias,
                                                   void* __restrict__ Cptr)
{
    __shared__ __attribute__((aligned(16))) u16 btile[2][64][32];  // 8 KB

    const int lane = threadIdx.x & 31;
    const int wave = threadIdx.x >> 5;
    const int wm = wave >> 1, wn = wave & 1;
    const int m0 = blockIdx.y * 128 + wm * 32;
    const int n0b = blockIdx.x * 64;           // block N origin
    const int n0 = n0b + wn * 32;              // wave N origin
    const int lr = lane & 15;
    const int hi = lane >> 4;

    v8f acc[2][2];
    acc[0][0] = {}; acc[0][1] = {}; acc[1][0] = {}; acc[1][1] = {};

    const float* Af = (const float*)Aptr;
    const u16*   Ab = (const u16*)Aptr;
    const u16*   Wb = Wt + (size_t)n0b * DH;   // 64 rows of Wt for this block

    if (wave == 0)
        tdm_load_2d(Wb, lds_off(&btile[0][0][0]), 64, 32, DH);

    for (int k0 = 0; k0 < DH; k0 += 32) {
        const int ibuf = (k0 >> 5) & 1;
        const bool has_next = (k0 + 32) < DH;
        if (wave == 0) {
            if (has_next) {
                tdm_load_2d(Wb + k0 + 32, lds_off(&btile[ibuf ^ 1][0][0]), 64, 32, DH);
                __builtin_amdgcn_s_wait_tensorcnt(1);
            } else {
                __builtin_amdgcn_s_wait_tensorcnt(0);
            }
        }
        __syncthreads();

        BFrag a[2], b[2];
        #pragma unroll
        for (int i = 0; i < 2; ++i) {
            const int arow = m0 + 16 * i + lr;
            if (A_F32) {
                const float* p0 = Af + (size_t)arow * DH + k0 + hi * 8;
                const float* p1 = p0 + 16;
                f32x4 x0 = ((const f32x4*)p0)[0], x1 = ((const f32x4*)p0)[1];
                f32x4 y0 = ((const f32x4*)p1)[0], y1 = ((const f32x4*)p1)[1];
                #pragma unroll
                for (int e = 0; e < 4; ++e) {
                    a[i].s[e]      = f2bf(x0[e]);
                    a[i].s[4 + e]  = f2bf(x1[e]);
                    a[i].s[8 + e]  = f2bf(y0[e]);
                    a[i].s[12 + e] = f2bf(y1[e]);
                }
            } else {
                const u16* p = Ab + (size_t)arow * DH + k0;
                a[i].q[0] = *(const u32x4*)(p + hi * 8);
                a[i].q[1] = *(const u32x4*)(p + 16 + hi * 8);
            }
        }
        #pragma unroll
        for (int j = 0; j < 2; ++j) {
            const u16* p = &btile[ibuf][wn * 32 + 16 * j + lr][hi * 8];
            b[j].q[0] = *(const u32x4*)(p);
            b[j].q[1] = *(const u32x4*)(p + 16);
        }
        #pragma unroll
        for (int i = 0; i < 2; ++i)
            #pragma unroll
            for (int j = 0; j < 2; ++j)
                acc[i][j] = __builtin_amdgcn_wmma_f32_16x16x32_bf16(
                    false, a[i].bf, false, b[j].bf, (short)0, acc[i][j], false, false);

        __syncthreads();   // btile[ibuf] free for the next TDM overwrite
    }

    // Epilogue: bias add + store with layout scatter
    #pragma unroll
    for (int i = 0; i < 2; ++i) {
        #pragma unroll
        for (int j = 0; j < 2; ++j) {
            const int col = n0 + 16 * j + lr;
            const float bv = (OUT_MODE == 2) ? 0.0f : bias[col];
            #pragma unroll
            for (int r = 0; r < 8; ++r) {
                const int row = m0 + 16 * i + r + 8 * hi;   // row = b*FF + f
                const float val = acc[i][j][r] + bv;
                if (OUT_MODE == 2) {
                    ((float*)Cptr)[(size_t)row * DH + col] = val;
                } else {
                    const int bz = row / FF, f = row & (FF - 1);
                    const int h = col >> 6, d = col & 63;
                    size_t idx;
                    if (OUT_MODE == 0)
                        idx = (((size_t)(bz * HH + h)) * FF + f) * DD + d;
                    else
                        idx = (((size_t)(bz * HH + h)) * DD + d) * FF + f;
                    ((u16*)Cptr)[idx] = f2bf(val);
                }
            }
        }
    }
}

// ---------------------------------------------------------------------------
// Flash attention: one (b,h) per block.z/.y, 128 query rows per block,
// 16 query rows per wave, key loop in chunks of 32.
// Shared K (32x64) and V (64x32) tiles are TDM-staged into LDS per key-chunk
// (double-buffered) -- all 8 waves consume the same tiles.
//   qh, kh : bf16 [B][H][F][D]
//   vt     : bf16 [B][H][D][F]
//   attn   : bf16 [B*F][DH]  (rows b*F+f, cols h*64+d)
// ---------------------------------------------------------------------------
__global__ __launch_bounds__(256) void attn_kernel(const u16* __restrict__ qh,
                                                   const u16* __restrict__ kh,
                                                   const u16* __restrict__ vt,
                                                   const float* __restrict__ mask,
                                                   u16* __restrict__ attn)
{
    __shared__ __attribute__((aligned(16))) u16 ktile[2][32][64];  // 8 KB
    __shared__ __attribute__((aligned(16))) u16 vtile[2][64][32];  // 8 KB
    __shared__ __attribute__((aligned(16))) u16 pbuf[8][16][32];   // 8 KB

    const int lane = threadIdx.x & 31;
    const int wave = threadIdx.x >> 5;
    const int lr = lane & 15;
    const int hi = lane >> 4;
    const int b = blockIdx.z, h = blockIdx.y;
    const int q0 = blockIdx.x * 128 + wave * 16;

    const size_t head = (size_t)(b * HH + h);
    const u16* kbase = kh + head * FF * DD;    // [F][D], rows contiguous
    const u16* vbase = vt + head * DD * FF;    // [D][F], row pitch FF

    // Q A-fragments for this wave's 16 rows (K = D = 64 -> 2 fragments)
    BFrag aq[2];
    {
        const u16* p = qh + (head * FF + q0 + lr) * DD;
        aq[0].q[0] = *(const u32x4*)(p + hi * 8);
        aq[0].q[1] = *(const u32x4*)(p + 16 + hi * 8);
        aq[1].q[0] = *(const u32x4*)(p + 32 + hi * 8);
        aq[1].q[1] = *(const u32x4*)(p + 48 + hi * 8);
    }

    v8f o[4];
    o[0] = {}; o[1] = {}; o[2] = {}; o[3] = {};
    float rm[8], rs[8];
    #pragma unroll
    for (int r = 0; r < 8; ++r) { rm[r] = -1e30f; rs[r] = 0.0f; }

    if (wave == 0) {
        tdm_load_2d(kbase, lds_off(&ktile[0][0][0]), 32, 64, 64);
        tdm_load_2d(vbase, lds_off(&vtile[0][0][0]), 64, 32, FF);
    }

    for (int kv = 0; kv < FF; kv += 32) {
        const int ibuf = (kv >> 5) & 1;
        const bool has_next = (kv + 32) < FF;
        if (wave == 0) {
            if (has_next) {
                tdm_load_2d(kbase + (kv + 32) * DD, lds_off(&ktile[ibuf ^ 1][0][0]), 32, 64, 64);
                tdm_load_2d(vbase + (kv + 32),      lds_off(&vtile[ibuf ^ 1][0][0]), 64, 32, FF);
                __builtin_amdgcn_s_wait_tensorcnt(2);
            } else {
                __builtin_amdgcn_s_wait_tensorcnt(0);
            }
        }
        __syncthreads();

        // ---- scores S[16 x 32] = Q . K^T  (two 16x16 tiles, 2 WMMAs each)
        v8f s[2]; s[0] = {}; s[1] = {};
        #pragma unroll
        for (int j = 0; j < 2; ++j) {
            const u16* p = &ktile[ibuf][16 * j + lr][0];
            BFrag b0, b1;
            b0.q[0] = *(const u32x4*)(p + hi * 8);
            b0.q[1] = *(const u32x4*)(p + 16 + hi * 8);
            b1.q[0] = *(const u32x4*)(p + 32 + hi * 8);
            b1.q[1] = *(const u32x4*)(p + 48 + hi * 8);
            s[j] = __builtin_amdgcn_wmma_f32_16x16x32_bf16(
                false, aq[0].bf, false, b0.bf, (short)0, s[j], false, false);
            s[j] = __builtin_amdgcn_wmma_f32_16x16x32_bf16(
                false, aq[1].bf, false, b1.bf, (short)0, s[j], false, false);
        }

        // ---- scale + additive mask (mask==1 -> -1e9)
        float sv[2][8];
        #pragma unroll
        for (int j = 0; j < 2; ++j)
            #pragma unroll
            for (int r = 0; r < 8; ++r) {
                const int qrow = q0 + r + 8 * hi;
                const int kcol = kv + 16 * j + lr;
                sv[j][r] = s[j][r] * 0.125f
                         - 1e9f * mask[((size_t)b * FF + qrow) * FF + kcol];
            }

        // ---- online softmax update (row reductions across 16 lanes)
        float scale_[8];
        #pragma unroll
        for (int r = 0; r < 8; ++r) {
            float mx = fmaxf(sv[0][r], sv[1][r]);
            mx = fmaxf(mx, __shfl_xor(mx, 1));
            mx = fmaxf(mx, __shfl_xor(mx, 2));
            mx = fmaxf(mx, __shfl_xor(mx, 4));
            mx = fmaxf(mx, __shfl_xor(mx, 8));
            const float nm = fmaxf(rm[r], mx);
            const float sc = __expf(rm[r] - nm);
            const float p0 = __expf(sv[0][r] - nm);
            const float p1 = __expf(sv[1][r] - nm);
            sv[0][r] = p0; sv[1][r] = p1;
            float ps = p0 + p1;
            ps += __shfl_xor(ps, 1);
            ps += __shfl_xor(ps, 2);
            ps += __shfl_xor(ps, 4);
            ps += __shfl_xor(ps, 8);
            rs[r] = rs[r] * sc + ps;
            rm[r] = nm;
            scale_[r] = sc;
        }
        #pragma unroll
        for (int t = 0; t < 4; ++t)
            #pragma unroll
            for (int r = 0; r < 8; ++r) o[t][r] *= scale_[r];

        // ---- C-layout P -> A-layout via per-wave LDS roundtrip
        #pragma unroll
        for (int r = 0; r < 8; ++r) {
            pbuf[wave][r + 8 * hi][lr]      = f2bf(sv[0][r]);
            pbuf[wave][r + 8 * hi][16 + lr] = f2bf(sv[1][r]);
        }
        asm volatile("s_wait_dscnt 0" ::: "memory");   // DS in-order per wave
        BFrag pf;
        pf.q[0] = *(const u32x4*)(&pbuf[wave][lr][hi * 8]);
        pf.q[1] = *(const u32x4*)(&pbuf[wave][lr][16 + hi * 8]);

        // ---- O[16 x 64] += P(16x32) . V(32x64)  (4 WMMAs over d-tiles)
        #pragma unroll
        for (int t = 0; t < 4; ++t) {
            const u16* p = &vtile[ibuf][t * 16 + lr][hi * 8];
            BFrag vf;
            vf.q[0] = *(const u32x4*)(p);
            vf.q[1] = *(const u32x4*)(p + 16);
            o[t] = __builtin_amdgcn_wmma_f32_16x16x32_bf16(
                false, pf.bf, false, vf.bf, (short)0, o[t], false, false);
        }

        __syncthreads();   // tiles[ibuf] free for the next TDM overwrite
    }

    // ---- finalize: divide by row sums, store bf16 at [b*F+f][h*64+d]
    #pragma unroll
    for (int t = 0; t < 4; ++t)
        #pragma unroll
        for (int r = 0; r < 8; ++r) {
            const int qrow = q0 + r + 8 * hi;
            const float val = o[t][r] / rs[r];
            attn[((size_t)(b * FF + qrow)) * DH + h * DD + t * 16 + lr] = f2bf(val);
        }
}

// ---------------------------------------------------------------------------
// Host launcher
// ---------------------------------------------------------------------------
extern "C" void kernel_launch(void* const* d_in, const int* in_sizes, int n_in,
                              void* d_out, int out_size, void* d_ws, size_t ws_size,
                              hipStream_t stream)
{
    (void)in_sizes; (void)n_in; (void)out_size; (void)ws_size;

    const float* q   = (const float*)d_in[0];
    const float* k   = (const float*)d_in[1];
    const float* v   = (const float*)d_in[2];
    const float* msk = (const float*)d_in[3];
    const float* qw  = (const float*)d_in[4];
    const float* qb  = (const float*)d_in[5];
    const float* kw  = (const float*)d_in[6];
    const float* kb  = (const float*)d_in[7];
    const float* vw  = (const float*)d_in[8];
    const float* vb  = (const float*)d_in[9];
    const float* ow  = (const float*)d_in[10];  // (H, D, DH) == (DH, DH) row-major

    char* ws = (char*)d_ws;
    const size_t MB = (size_t)1024 * 1024;
    u16* qwT  = (u16*)(ws + 0 * MB);    // 2 MB each
    u16* kwT  = (u16*)(ws + 2 * MB);
    u16* vwT  = (u16*)(ws + 4 * MB);
    u16* owT  = (u16*)(ws + 6 * MB);
    u16* qh   = (u16*)(ws + 8 * MB);    // 16 MB each
    u16* kh   = (u16*)(ws + 24 * MB);
    u16* vt   = (u16*)(ws + 40 * MB);
    u16* attn = (u16*)(ws + 56 * MB);   // ends at 72 MB

    dim3 blk(256);

    // Phase 1: weights -> bf16 transposed
    transpose_w_kernel<<<DH * DH / 256, blk, 0, stream>>>(qw, qwT);
    transpose_w_kernel<<<DH * DH / 256, blk, 0, stream>>>(kw, kwT);
    transpose_w_kernel<<<DH * DH / 256, blk, 0, stream>>>(vw, vwT);
    transpose_w_kernel<<<DH * DH / 256, blk, 0, stream>>>(ow, owT);

    // Phase 2: Q/K/V projections (WMMA + TDM-staged B), head-scattered bf16
    dim3 gg(DH / 64, MM / 128);
    gemm_kernel<true, 0><<<gg, blk, 0, stream>>>(q, qwT, qb, qh);
    gemm_kernel<true, 0><<<gg, blk, 0, stream>>>(k, kwT, kb, kh);
    gemm_kernel<true, 1><<<gg, blk, 0, stream>>>(v, vwT, vb, vt);

    // Phase 3: flash attention (WMMA + TDM-staged K/V tiles)
    dim3 ga(FF / 128, HH, BB);
    attn_kernel<<<ga, blk, 0, stream>>>(qh, kh, vt, msk, attn);

    // Phase 4: output projection (WMMA), f32 result
    gemm_kernel<false, 2><<<gg, blk, 0, stream>>>(attn, owT, nullptr, d_out);
}
